// CustomLSTM_70738111365197
// MI455X (gfx1250) — matrix operand order repaired and checked by reading, not compiled, not run on hardware
//
#include <hip/hip_runtime.h>

typedef __attribute__((ext_vector_type(16))) __bf16 v16bf;
typedef __attribute__((ext_vector_type(8)))  float  v8f;

// Global-address-space pointer types: restore global_load codegen after the
// opaque asm round-trip (which otherwise degrades loads to FLAT).
typedef const __bf16 __attribute__((address_space(1)))* gbfp;
typedef const v16bf  __attribute__((address_space(1)))* gv16p;

#define HID   256
#define TLEN  720
#define DIN   48
#define BROWS 16            // batch rows per workgroup
#define LSTRIDE 264         // padded LDS row stride (bf16 elems)

// ---------------- helpers ----------------

__device__ __forceinline__ float sigf(float x) {
    return 1.0f / (1.0f + __expf(-x));
}
__device__ __forceinline__ float tanh_fast(float x) {
    return 2.0f * sigf(2.0f * x) - 1.0f;   // tanh(x) = 2*sigmoid(2x) - 1
}

// Load a 16x32 bf16 A-fragment from LDS in contiguous K order (matches B pack):
// lane L holds row M=L&15; lanes 16-31 take K offset by 16;
// VGPR i -> K = i*2 (+halfSel*16).
// Net per-lane access = one contiguous 32B chunk -> 2x ds_load_b128.
__device__ __forceinline__ v16bf load_a_frag(const __bf16* p, int stride, int lane) {
    const int m  = lane & 15;
    const int hs = (lane >> 4) * 16;
    const __bf16* row = p + m * stride + hs;
    v16bf a;
#pragma unroll
    for (int i = 0; i < 8; ++i) {
        const int klo = i * 2;
        a[2 * i]     = row[klo];
        a[2 * i + 1] = row[klo + 1];
    }
    return a;
}

// B-fragments were pre-swizzled so each lane reads 32 contiguous bytes
// (global address space -> 2x global_load_b128, SADDR + imm offset form).
__device__ __forceinline__ v16bf load_b_frag(gbfp frag_base, int lane) {
    return *(gv16p)(frag_base + lane * 16);
}

// ---------------- kernel 0: weight repack to WMMA B-fragment layout ----------------
// Gates B matrix: K = 512 (0..255 = emb-side via ih, 256..511 = h-side via hh),
// N = 1024 (4 gates x 256).  Reference multiplies e @ x_i with NO transpose, so
// W[k][g*256+j] = ih[g*256 + k][j]  (and hh likewise).
// Fragment f = nt*16 + kb stores 32 lanes x 16 bf16, lane l value j:
//   k = kb*32 + (l>=16 ? 16 : 0) + j ; n = nt*16 + (l&15)
__global__ void lstm_repack_kernel(const float* __restrict__ ih,
                                   const float* __restrict__ hh,
                                   const float* __restrict__ emb_w,
                                   __bf16* __restrict__ wpack,
                                   __bf16* __restrict__ wemb) {
    const int idx = blockIdx.x * blockDim.x + threadIdx.x;
    const int NG = 512 * 1024;
    if (idx < NG) {
        const int j  = idx & 15;
        const int l  = (idx >> 4) & 31;
        const int f  = idx >> 9;
        const int kb = f & 15;
        const int nt = f >> 4;
        const int k  = kb * 32 + ((l >= 16) ? 16 : 0) + j;
        const int n  = nt * 16 + (l & 15);
        const int g  = n >> 8;
        const int cc = n & 255;
        float v;
        if (k < 256) v = ih[(g * 256 + k) * 256 + cc];
        else         v = hh[(g * 256 + (k - 256)) * 256 + cc];
        wpack[idx] = (__bf16)v;
    } else {
        const int idx2 = idx - NG;           // embedding B: K padded 48->64, N=256
        if (idx2 < 64 * 256) {
            const int j  = idx2 & 15;
            const int l  = (idx2 >> 4) & 31;
            const int f  = idx2 >> 9;        // f = nt*2 + kb
            const int kb = f & 1;
            const int nt = f >> 1;
            const int k  = kb * 32 + ((l >= 16) ? 16 : 0) + j;
            const int n  = nt * 16 + (l & 15);
            // e = x @ emb_w^T  =>  B[k][n] = emb_w[n][k]; pad K 48..63 with 0
            const float v = (k < DIN) ? emb_w[n * DIN + k] : 0.0f;
            wemb[idx2] = (__bf16)v;
        }
    }
}

// ---------------- kernel 1: fused persistent LSTM ----------------
// 16 workgroups (one per 16-row batch tile), 512 threads = 16 wave32 waves.
// Wave w owns h-columns [w*16, w*16+16); its 4 gate tiles are at
// n-tiles nt = g*16 + w.  c state lives in that wave's registers forever.
// Weights stay L2-resident (1 MB total vs 192 MB L2); we deliberately re-read
// the packed B-fragments from global each step (opaque pointer defeats LICM so
// the compiler cannot hoist ~2KB/lane of fragments and spill them to scratch).
__global__ __launch_bounds__(512, 1)
void lstm_persistent_kernel(const float* __restrict__ x,
                            const float* __restrict__ emb_b,
                            const float* __restrict__ ib,
                            const float* __restrict__ hb,
                            const float* __restrict__ head_w,
                            const float* __restrict__ head_b,
                            const __bf16* __restrict__ wpack,
                            const __bf16* __restrict__ wemb,
                            float* __restrict__ out) {
    __shared__ __bf16 h_lds[BROWS][LSTRIDE];
    __shared__ __bf16 e_lds[BROWS][LSTRIDE];
    __shared__ __bf16 x_lds[BROWS][64];
    __shared__ float  hpart[16][2][16];

    const int tid  = threadIdx.x;
    const int w    = tid >> 5;
    const int lane = tid & 31;
    const int b0   = blockIdx.x * BROWS;

    const int nloc    = lane & 15;
    const int halfSel = lane >> 4;
    const int col     = w * 16 + nloc;      // this lane's output column

    float bias[4];
#pragma unroll
    for (int g = 0; g < 4; ++g) bias[g] = ib[g * 256 + col] + hb[g * 256 + col];
    const float ebias = emb_b[col];
    const float hbias = head_b[0];

    // h0 = 0
    for (int i = tid; i < BROWS * LSTRIDE; i += 512)
        ((__bf16*)h_lds)[i] = (__bf16)0.0f;

    v8f c_reg = {0.f, 0.f, 0.f, 0.f, 0.f, 0.f, 0.f, 0.f};   // c0 = 0
    __syncthreads();

    for (int t = 0; t < TLEN; ++t) {
        // Opaque copies: force the weight base pointers to be redefined every
        // iteration so fragment loads stay inside the loop (L2 hits) instead
        // of being hoisted + spilled to scratch.  The subsequent explicit
        // addrspace(1) cast restores global_load (SADDR) codegen.
        const __bf16* wp_ = wpack;
        const __bf16* we_ = wemb;
        asm volatile("" : "+s"(wp_), "+s"(we_));
        gbfp wp = (gbfp)wp_;
        gbfp we = (gbfp)we_;

        // ---- stage x[btile, t, :] into LDS as bf16, K padded to 64 ----
        for (int i = tid; i < BROWS * 64; i += 512) {
            const int r = i >> 6, d = i & 63;
            const float v = (d < DIN) ? x[((b0 + r) * TLEN + t) * DIN + d] : 0.0f;
            x_lds[r][d] = (__bf16)v;
        }
        // prefetch next timestep's x rows into L2/L0 (global_prefetch_b8)
        if (t + 1 < TLEN && tid < BROWS) {
            __builtin_prefetch(&x[((b0 + tid) * TLEN + (t + 1)) * DIN], 0, 1);
        }
        __syncthreads();

        // ---- embedding tile: wave w computes e[:, w*16 .. w*16+16) ----
        {
            v8f acc;
#pragma unroll
            for (int i = 0; i < 8; ++i) acc[i] = ebias;
#pragma unroll
            for (int kb = 0; kb < 2; ++kb) {
                v16bf a = load_a_frag(&x_lds[0][kb * 32], 64, lane);
                v16bf b = load_b_frag(we + ((w * 2 + kb) << 9), lane);
                acc = __builtin_amdgcn_wmma_f32_16x16x32_bf16(
                          false, a, false, b, (short)0, acc, false, false);
            }
#pragma unroll
            for (int r = 0; r < 8; ++r) {
                const int m = r + halfSel * 8;      // D layout: lanes 16-31 -> M+8
                e_lds[m][col] = (__bf16)acc[r];
            }
        }
        __syncthreads();

        // ---- gates: [e | h] (16x512) @ Wpack(512x1024), this wave's 4 tiles ----
        v8f accI, accF, accG, accO;
#pragma unroll
        for (int i = 0; i < 8; ++i) {
            accI[i] = bias[0]; accF[i] = bias[1];
            accG[i] = bias[2]; accO[i] = bias[3];
        }
#pragma unroll
        for (int kb = 0; kb < 16; ++kb) {
            v16bf a = (kb < 8) ? load_a_frag(&e_lds[0][kb * 32], LSTRIDE, lane)
                               : load_a_frag(&h_lds[0][(kb - 8) * 32], LSTRIDE, lane);
            v16bf bI = load_b_frag(wp + ((( 0 + w) * 16 + kb) << 9), lane);
            v16bf bF = load_b_frag(wp + (((16 + w) * 16 + kb) << 9), lane);
            v16bf bG = load_b_frag(wp + (((32 + w) * 16 + kb) << 9), lane);
            v16bf bO = load_b_frag(wp + (((48 + w) * 16 + kb) << 9), lane);
            accI = __builtin_amdgcn_wmma_f32_16x16x32_bf16(false, a, false, bI, (short)0, accI, false, false);
            accF = __builtin_amdgcn_wmma_f32_16x16x32_bf16(false, a, false, bF, (short)0, accF, false, false);
            accG = __builtin_amdgcn_wmma_f32_16x16x32_bf16(false, a, false, bG, (short)0, accG, false, false);
            accO = __builtin_amdgcn_wmma_f32_16x16x32_bf16(false, a, false, bO, (short)0, accO, false, false);
        }
        __syncthreads();   // all reads of e_lds/h_lds done before overwriting h

        // ---- cell update (registers only) + publish h to LDS ----
#pragma unroll
        for (int r = 0; r < 8; ++r) {
            const float I = accI[r], F = accF[r], G = accG[r], O = accO[r];
            const float c1 = c_reg[r] * sigf(F) + sigf(I) * tanh_fast(G);
            const float h1 = sigf(O) + tanh_fast(c1);   // faithful: add, not mul
            c_reg[r] = c1;
            const int m = r + halfSel * 8;
            h_lds[m][col] = (__bf16)h1;
        }
        __syncthreads();

        // ---- head: out[b,t] = h . head_w + head_b (deterministic reduction) ----
        {
            const int r = nloc;
            const int cbase = w * 16 + halfSel * 8;
            float s = 0.f;
#pragma unroll
            for (int c2 = 0; c2 < 8; ++c2)
                s += (float)h_lds[r][cbase + c2] * head_w[cbase + c2];
            hpart[w][halfSel][r] = s;
        }
        __syncthreads();
        if (tid < 16) {
            float s = hbias;
#pragma unroll
            for (int ww = 0; ww < 16; ++ww)
                s += hpart[ww][0][tid] + hpart[ww][1][tid];
            out[(b0 + tid) * TLEN + t] = s;
        }
        // next iteration's x_lds store is safe: all x_lds readers passed 4 barriers
    }
}

// ---------------- host launcher ----------------
extern "C" void kernel_launch(void* const* d_in, const int* in_sizes, int n_in,
                              void* d_out, int out_size, void* d_ws, size_t ws_size,
                              hipStream_t stream) {
    const float* x      = (const float*)d_in[0];
    const float* emb_w  = (const float*)d_in[1];
    const float* emb_b  = (const float*)d_in[2];
    const float* ih     = (const float*)d_in[3];
    const float* ib     = (const float*)d_in[4];
    const float* hh     = (const float*)d_in[5];
    const float* hb     = (const float*)d_in[6];
    const float* head_w = (const float*)d_in[7];
    const float* head_b = (const float*)d_in[8];
    float* out = (float*)d_out;

    __bf16* wpack = (__bf16*)d_ws;                 // 512*1024 bf16 = 1 MB
    __bf16* wemb  = wpack + 512 * 1024;            // 64*256 bf16  = 32 KB

    // 524288 + 16384 = 540672 elements = 2112 blocks of 256
    lstm_repack_kernel<<<2112, 256, 0, stream>>>(ih, hh, emb_w, wpack, wemb);

    // 16 batch tiles x 512 threads (16 wave32 waves)
    lstm_persistent_kernel<<<16, 512, 0, stream>>>(
        x, emb_b, ib, hb, head_w, head_b, wpack, wemb, out);
}
